// EnsembleRSSM_16664473108887
// MI455X (gfx1250) — matrix operand order repaired and checked by reading, not compiled, not run on hardware
//
#include <hip/hip_runtime.h>
#include <stdint.h>

// ---------------- types ----------------
typedef __attribute__((ext_vector_type(16))) __bf16 v16bf;
typedef __attribute__((ext_vector_type(8)))  float  v8f;

#define B_     1024
#define T_     64
#define EMBED_ 1024
#define ACT_   32
#define STOCH_ 30
#define DETER_ 200
#define HIDDEN_ 200
#define OUTC_  380

// padded GEMM dims (K mult of 32, N mult of 16); concat segments padded to 16-multiples
#define K_A0   64    // stoch(30)+act(32) = 62 -> 64          (2 kt)
#define K_CAT1 416   // x pad 208 | deter pad 208             (13 kt)
#define K_DET  224   // 200 -> 224                            (7 kt)
#define K_CAT2 1248  // deter pad 208 | embed 1024 | pad 16   (39 kt)
#define NT_HID 13    // 200 -> 208
#define NT_GRU 38    // 600 -> 608
#define NT_DST 4     // 60  -> 64

#define TILES_WII  (2*NT_HID)      // 26
#define TILES_WGRU (13*NT_GRU)     // 494
#define TILES_WIO  (7*NT_HID)      // 91  (per ensemble member)
#define TILES_WID  (7*NT_DST)      // 28  (per member)
#define TILES_WOO  (39*NT_HID)     // 507
#define TILES_WOD  (7*NT_DST)      // 28

__device__ __forceinline__ unsigned short f2bfu(float f) {
    union { float f; unsigned u; } x; x.f = f;
    unsigned r = x.u + 0x7FFFu + ((x.u >> 16) & 1u);   // round-to-nearest-even
    return (unsigned short)(r >> 16);
}

// branchless activations on hardware transcendentals (no exec divergence)
__device__ __forceinline__ float elu_f(float x) {
    float e = __expf(fminf(x, 0.f)) - 1.f;
    return x > 0.f ? x : e;
}
__device__ __forceinline__ float sigm_f(float x) {
    return __builtin_amdgcn_rcpf(1.f + __expf(-x));
}
__device__ __forceinline__ float softplus_f(float x) {
    float e  = __expf(fminf(x, 20.f));
    float sp = __logf(1.f + e);
    return x > 20.f ? x : sp;
}
__device__ __forceinline__ float tanh_f(float x) {
    return 1.f - 2.f * __builtin_amdgcn_rcpf(__expf(2.f * x) + 1.f);
}

// ---------------- weight prep: f32 [K x N] row-major -> bf16 WMMA B-fragment tiles ----------------
// Split-K remap: padded row k maps to source row  k<K1 ? k : (K1pad<=k<K1pad+K2 ? K1+(k-K1pad) : zero)
// tile (kt,nt): 32 lanes x 16 bf16, lane-major. lane L: n = nt*16 + L%16, k = kt*32 + (L/16)*8 + (p&7) + (p>>3)*16
__global__ void prep_tiles(const float* __restrict__ src, unsigned short* __restrict__ dst,
                           int K1, int K1pad, int K2, int N, int NT) {
    int tile = blockIdx.x;
    int kt = tile / NT, nt = tile % NT;
    int e0 = threadIdx.x * 2;
#pragma unroll
    for (int q = 0; q < 2; ++q) {
        int e = e0 + q;
        int lane = e >> 4, p = e & 15;
        int v = p >> 1, j = p & 1;
        int n = nt * 16 + (lane & 15);
        int k = kt * 32 + (lane >> 4) * 8 + ((v * 2 + j) & 7) + ((v * 2 + j) >> 3) * 16;
        int ksrc = (k < K1) ? k : ((k >= K1pad && k < K1pad + K2) ? K1 + (k - K1pad) : -1);
        float val = (ksrc >= 0 && n < N) ? src[(size_t)ksrc * N + n] : 0.f;
        dst[(size_t)tile * 512 + e] = f2bfu(val);
    }
}

// ---------------- fixed-count multi-tile WMMA GEMM ----------------
// CNT is compile-time: the kt-loop body is unconditional, so accumulators keep
// their registers (wmma dst==src2, no phi copy storms). kt loop stays rolled.
template <int CNT>
__device__ __forceinline__ void wmma_gemm_fixed(const unsigned short* actBase, int kStride,
                                                const unsigned short* wt, int NT, int nt0,
                                                int KT, int lane, v8f* acc) {
    const int m = lane & 15, half = lane >> 4;
#pragma unroll
    for (int i = 0; i < CNT; ++i) acc[i] = (v8f){};
#pragma unroll 1
    for (int kt = 0; kt < KT; ++kt) {
        union { v16bf v; uint4 q[2]; } a;
        const unsigned short* ap = actBase + m * kStride + kt * 32 + half * 8;
        a.q[0] = *(const uint4*)(ap);        // v0..v3: k = base + 0..7
        a.q[1] = *(const uint4*)(ap + 16);   // v4..v7: k = base + 16..23
#pragma unroll
        for (int i = 0; i < CNT; ++i) {
            union { v16bf v; uint4 q[2]; } b;
            const uint4* bp = (const uint4*)(wt + (size_t)(kt * NT + nt0 + i) * 512) + lane * 2;
            b.q[0] = bp[0];
            b.q[1] = bp[1];
            acc[i] = __builtin_amdgcn_wmma_f32_16x16x32_bf16(false, a.v, false, b.v,
                                                             (short)0, acc[i], false, false);
        }
    }
}

// ---------------- main persistent RSSM scan kernel ----------------
__global__ __launch_bounds__(256)
void rssm_scan_kernel(const float* __restrict__ embed, const float* __restrict__ action,
                      const unsigned char* __restrict__ is_first,
                      const float* __restrict__ eps_prior, const float* __restrict__ eps_post,
                      const int* __restrict__ idx,
                      const float* __restrict__ b_img_in, const float* __restrict__ b_gru,
                      const float* __restrict__ ln_scale, const float* __restrict__ ln_bias,
                      const float* __restrict__ b_img_out, const float* __restrict__ b_img_dist,
                      const float* __restrict__ b_obs_out, const float* __restrict__ b_obs_dist,
                      const unsigned short* __restrict__ gWii, const unsigned short* __restrict__ gWgru,
                      const unsigned short* __restrict__ gWio, const unsigned short* __restrict__ gWid,
                      const unsigned short* __restrict__ gWoo, const unsigned short* __restrict__ gWod,
                      float* __restrict__ out) {
    __shared__ unsigned short A0[16][K_A0];        // [stoch(30) | act(32) | pad]
    __shared__ unsigned short cat1[16][K_CAT1];    // [x 0..207 | deter 208..407 | pad]
    __shared__ float          parts[16][608];      // GRU pre-activations f32
    __shared__ unsigned short dbf[16][K_DET];      // new deter bf16 (img_out A)
    __shared__ unsigned short cat2[16][K_CAT2];    // [deter 0..207 | embed 208..1231 | pad]
    __shared__ unsigned short hbf[16][K_DET];      // elu(img_out) bf16
    __shared__ unsigned short hobf[16][K_DET];     // elu(obs_out) bf16
    __shared__ float          dtmp[16][64];        // dist head scratch f32
    __shared__ float          s_stoch[16][STOCH_];
    __shared__ float          s_deter[16][DETER_];

    const int tid = threadIdx.x;
    const int lane = tid & 31;
    const int wv = __builtin_amdgcn_readfirstlane(tid >> 5);   // scalar wave id
    const int b0 = blockIdx.x * 16;
    const int col0 = lane & 15, mrow0 = (lane >> 4) * 8;

    // wave-uniform tile schedules (SGPR values -> scalar branches, SGPR-based addressing)
    const int nt13_0 = __builtin_amdgcn_readfirstlane((wv < 5) ? 2 * wv : 10 + (wv - 5));
    const int nt13_c = __builtin_amdgcn_readfirstlane((wv < 5) ? 2 : 1);
    const int nt38_0 = __builtin_amdgcn_readfirstlane((wv < 6) ? 5 * wv : 30 + 4 * (wv - 6));
    const int nt38_c = __builtin_amdgcn_readfirstlane((wv < 6) ? 5 : 4);

    // zero state + zero padded columns once (never written afterwards)
    for (int i = tid; i < 16 * STOCH_; i += 256) s_stoch[i / STOCH_][i % STOCH_] = 0.f;
    for (int i = tid; i < 16 * DETER_; i += 256) s_deter[i / DETER_][i % DETER_] = 0.f;
    for (int i = tid; i < 16 * 2;  i += 256) A0[i / 2][62 + (i & 1)] = 0;
    for (int i = tid; i < 16 * 8;  i += 256) cat1[i / 8][408 + (i & 7)] = 0;
    for (int i = tid; i < 16 * 24; i += 256) {
        int m = i / 24, j = i % 24;
        dbf[m][200 + j] = 0;
        if (j < 16) { hbf[m][208 + j] = 0; hobf[m][208 + j] = 0; cat2[m][1232 + j] = 0; }
        if (j < 8)  cat2[m][200 + j] = 0;
    }
    __syncthreads();

    for (int t = 0; t < T_; ++t) {
        const int kmem = __builtin_amdgcn_readfirstlane(idx[t]);

        // ---- stage 0: mask state, build A0 and deter-half of cat1; load embed into cat2 ----
        for (int i = tid; i < 16 * DETER_; i += 256) {
            int m = i / DETER_, j = i % DETER_;
            float msk = is_first[(size_t)(b0 + m) * T_ + t] ? 0.f : 1.f;
            float d = s_deter[m][j] * msk;
            s_deter[m][j] = d;
            cat1[m][208 + j] = f2bfu(d);
        }
        for (int i = tid; i < 16 * STOCH_; i += 256) {
            int m = i / STOCH_, j = i % STOCH_;
            float msk = is_first[(size_t)(b0 + m) * T_ + t] ? 0.f : 1.f;
            A0[m][j] = f2bfu(s_stoch[m][j] * msk);
        }
        for (int i = tid; i < 16 * ACT_; i += 256) {
            int m = i / ACT_, j = i % ACT_;
            float msk = is_first[(size_t)(b0 + m) * T_ + t] ? 0.f : 1.f;
            A0[m][STOCH_ + j] = f2bfu(action[((size_t)(b0 + m) * T_ + t) * ACT_ + j] * msk);
        }
        for (int i = tid; i < 16 * EMBED_; i += 256) {
            int m = i >> 10, j = i & 1023;
            cat2[m][208 + j] = f2bfu(embed[((size_t)(b0 + m) * T_ + t) * EMBED_ + j]);
        }
        if (t + 1 < T_) {  // prefetch next step's embed rows (cacheline-strided)
            int m = tid >> 4, c = tid & 15;
            __builtin_prefetch(&embed[((size_t)(b0 + m) * T_ + (t + 1)) * EMBED_ + c * 64], 0, 0);
        }
        __syncthreads();

        // ---- GEMM1: x = elu(A0 @ W_img_in + b) -> cat1[:, 0:208] (pad cols killed by zero W rows) ----
        {
            v8f acc[2];
            if (nt13_c == 2) wmma_gemm_fixed<2>(&A0[0][0], K_A0, gWii, NT_HID, nt13_0, 2, lane, acc);
            else             wmma_gemm_fixed<1>(&A0[0][0], K_A0, gWii, NT_HID, nt13_0, 2, lane, acc);
#pragma unroll
            for (int i = 0; i < 2; ++i) {
                if (i < nt13_c) {
                    int col = (nt13_0 + i) * 16 + col0;
                    float bv = b_img_in[col < HIDDEN_ ? col : HIDDEN_ - 1];
#pragma unroll
                    for (int r = 0; r < 8; ++r)
                        cat1[mrow0 + r][col] = f2bfu(elu_f(acc[i][r] + bv));
                }
            }
        }
        __syncthreads();

        // ---- GEMM2: parts = cat1 @ W_gru + b_gru ----
        {
            v8f acc[5];
            if (nt38_c == 5) wmma_gemm_fixed<5>(&cat1[0][0], K_CAT1, gWgru, NT_GRU, nt38_0, 13, lane, acc);
            else             wmma_gemm_fixed<4>(&cat1[0][0], K_CAT1, gWgru, NT_GRU, nt38_0, 13, lane, acc);
#pragma unroll
            for (int i = 0; i < 5; ++i) {
                if (i < nt38_c) {
                    int col = (nt38_0 + i) * 16 + col0;
                    float bv = b_gru[col < 600 ? col : 599];
#pragma unroll
                    for (int r = 0; r < 8; ++r)
                        parts[mrow0 + r][col] = acc[i][r] + bv;
                }
            }
        }
        __syncthreads();

        // ---- LayerNorm(600) + GRU gates; one wave per 2 rows ----
        for (int m = wv * 2; m < wv * 2 + 2; ++m) {
            float sum = 0.f, sq = 0.f;
            for (int i = lane; i < 600; i += 32) { float p = parts[m][i]; sum += p; sq += p * p; }
#pragma unroll
            for (int o = 16; o > 0; o >>= 1) {
                sum += __shfl_xor(sum, o, 32);
                sq  += __shfl_xor(sq,  o, 32);
            }
            float mean = sum * (1.f / 600.f);
            float var  = sq * (1.f / 600.f) - mean * mean;
            float rstd = rsqrtf(var + 1e-5f);
            for (int j = lane; j < DETER_; j += 32) {
                float pr = (parts[m][j      ] - mean) * rstd * ln_scale[j      ] + ln_bias[j      ];
                float pc = (parts[m][j + 200] - mean) * rstd * ln_scale[j + 200] + ln_bias[j + 200];
                float pu = (parts[m][j + 400] - mean) * rstd * ln_scale[j + 400] + ln_bias[j + 400];
                float rg = sigm_f(pr);
                float cg = tanh_f(rg * pc);
                float ug = sigm_f(pu - 1.f);            // UPDATE_BIAS = -1
                float dn = ug * cg + (1.f - ug) * s_deter[m][j];
                s_deter[m][j] = dn;
                unsigned short hb = f2bfu(dn);
                dbf[m][j] = hb;
                cat2[m][j] = hb;
                out[((size_t)(b0 + m) * T_ + t) * OUTC_ + 180 + j] = dn;
            }
        }
        __syncthreads();

        // ---- GEMM3: h = elu(deter @ W_img_out[k] + b) ----
        {
            const unsigned short* wio = gWio + (size_t)kmem * TILES_WIO * 512;
            v8f acc[2];
            if (nt13_c == 2) wmma_gemm_fixed<2>(&dbf[0][0], K_DET, wio, NT_HID, nt13_0, 7, lane, acc);
            else             wmma_gemm_fixed<1>(&dbf[0][0], K_DET, wio, NT_HID, nt13_0, 7, lane, acc);
#pragma unroll
            for (int i = 0; i < 2; ++i) {
                if (i < nt13_c) {
                    int col = (nt13_0 + i) * 16 + col0;
                    float bv = b_img_out[kmem * HIDDEN_ + (col < HIDDEN_ ? col : HIDDEN_ - 1)];
#pragma unroll
                    for (int r = 0; r < 8; ++r)
                        hbf[mrow0 + r][col] = f2bfu(elu_f(acc[i][r] + bv));
                }
            }
        }
        __syncthreads();

        // ---- GEMM4: prior suff-stats = h @ W_img_dist[k] + b ----
        if (wv < NT_DST) {
            const unsigned short* wid = gWid + (size_t)kmem * TILES_WID * 512;
            v8f acc[1];
            wmma_gemm_fixed<1>(&hbf[0][0], K_DET, wid, NT_DST, wv, 7, lane, acc);
            int col = wv * 16 + col0;
            float bv = b_img_dist[kmem * 60 + (col < 60 ? col : 59)];
#pragma unroll
            for (int r = 0; r < 8; ++r)
                dtmp[mrow0 + r][col] = acc[0][r] + bv;
        }
        __syncthreads();
        for (int i = tid; i < 16 * STOCH_; i += 256) {   // prior fixup
            int m = i / STOCH_, j = i % STOCH_;
            float pm = dtmp[m][j];
            float ps = softplus_f(dtmp[m][j + 30]) + 0.1f;
            float pri = pm + ps * eps_prior[((size_t)t * B_ + (b0 + m)) * STOCH_ + j];
            size_t base = ((size_t)(b0 + m) * T_ + t) * OUTC_;
            out[base + 90 + j] = pm; out[base + 120 + j] = ps; out[base + 150 + j] = pri;
        }
        __syncthreads();

        // ---- GEMM5: ho = elu(cat(deter, embed) @ W_obs_out + b) ----
        {
            v8f acc[2];
            if (nt13_c == 2) wmma_gemm_fixed<2>(&cat2[0][0], K_CAT2, gWoo, NT_HID, nt13_0, 39, lane, acc);
            else             wmma_gemm_fixed<1>(&cat2[0][0], K_CAT2, gWoo, NT_HID, nt13_0, 39, lane, acc);
#pragma unroll
            for (int i = 0; i < 2; ++i) {
                if (i < nt13_c) {
                    int col = (nt13_0 + i) * 16 + col0;
                    float bv = b_obs_out[col < HIDDEN_ ? col : HIDDEN_ - 1];
#pragma unroll
                    for (int r = 0; r < 8; ++r)
                        hobf[mrow0 + r][col] = f2bfu(elu_f(acc[i][r] + bv));
                }
            }
        }
        __syncthreads();

        // ---- GEMM6: posterior suff-stats = ho @ W_obs_dist + b ----
        if (wv < NT_DST) {
            v8f acc[1];
            wmma_gemm_fixed<1>(&hobf[0][0], K_DET, gWod, NT_DST, wv, 7, lane, acc);
            int col = wv * 16 + col0;
            float bv = b_obs_dist[col < 60 ? col : 59];
#pragma unroll
            for (int r = 0; r < 8; ++r)
                dtmp[mrow0 + r][col] = acc[0][r] + bv;
        }
        __syncthreads();
        for (int i = tid; i < 16 * STOCH_; i += 256) {   // posterior fixup + next stoch state
            int m = i / STOCH_, j = i % STOCH_;
            float om  = dtmp[m][j];
            float osd = softplus_f(dtmp[m][j + 30]) + 0.1f;
            float po  = om + osd * eps_post[((size_t)t * B_ + (b0 + m)) * STOCH_ + j];
            size_t base = ((size_t)(b0 + m) * T_ + t) * OUTC_;
            out[base + j] = om; out[base + 30 + j] = osd; out[base + 60 + j] = po;
            s_stoch[m][j] = po;
        }
        __syncthreads();
    }
}

extern "C" void kernel_launch(void* const* d_in, const int* in_sizes, int n_in,
                              void* d_out, int out_size, void* d_ws, size_t ws_size,
                              hipStream_t stream) {
    (void)in_sizes; (void)n_in; (void)out_size; (void)ws_size;
    const float* embed      = (const float*)d_in[0];
    const float* action     = (const float*)d_in[1];
    const unsigned char* is_first = (const unsigned char*)d_in[2];
    const float* eps_prior  = (const float*)d_in[3];
    const float* eps_post   = (const float*)d_in[4];
    const int*   idx        = (const int*)d_in[5];
    const float* W_img_in   = (const float*)d_in[6];
    const float* b_img_in   = (const float*)d_in[7];
    const float* W_gru      = (const float*)d_in[8];
    const float* b_gru      = (const float*)d_in[9];
    const float* ln_scale   = (const float*)d_in[10];
    const float* ln_bias    = (const float*)d_in[11];
    const float* W_img_out  = (const float*)d_in[12];
    const float* b_img_out  = (const float*)d_in[13];
    const float* W_img_dist = (const float*)d_in[14];
    const float* b_img_dist = (const float*)d_in[15];
    const float* W_obs_out  = (const float*)d_in[16];
    const float* b_obs_out  = (const float*)d_in[17];
    const float* W_obs_dist = (const float*)d_in[18];
    const float* b_obs_dist = (const float*)d_in[19];
    float* out = (float*)d_out;

    unsigned short* ws   = (unsigned short*)d_ws;
    unsigned short* gWii  = ws;
    unsigned short* gWgru = gWii  + (size_t)TILES_WII  * 512;
    unsigned short* gWio  = gWgru + (size_t)TILES_WGRU * 512;
    unsigned short* gWid  = gWio  + (size_t)5 * TILES_WIO * 512;
    unsigned short* gWoo  = gWid  + (size_t)5 * TILES_WID * 512;
    unsigned short* gWod  = gWoo  + (size_t)TILES_WOO * 512;

    // convert + tile weights to bf16 fragment layout (~1.7 MB, L2-resident)
    prep_tiles<<<TILES_WII,  256, 0, stream>>>(W_img_in, gWii, 62, 62, 0, 200, NT_HID);
    prep_tiles<<<TILES_WGRU, 256, 0, stream>>>(W_gru, gWgru, 200, 208, 200, 600, NT_GRU);   // x|deter split
    for (int m = 0; m < 5; ++m) {
        prep_tiles<<<TILES_WIO, 256, 0, stream>>>(W_img_out  + (size_t)m * 200 * 200,
                                                  gWio + (size_t)m * TILES_WIO * 512, 200, 200, 0, 200, NT_HID);
        prep_tiles<<<TILES_WID, 256, 0, stream>>>(W_img_dist + (size_t)m * 200 * 60,
                                                  gWid + (size_t)m * TILES_WID * 512, 200, 200, 0, 60, NT_DST);
    }
    prep_tiles<<<TILES_WOO, 256, 0, stream>>>(W_obs_out, gWoo, 200, 208, 1024, 200, NT_HID); // deter|embed split
    prep_tiles<<<TILES_WOD, 256, 0, stream>>>(W_obs_dist, gWod, 200, 200, 0, 60, NT_DST);

    // persistent scan: 64 blocks x 16 batch rows, all 64 timesteps in-kernel
    rssm_scan_kernel<<<B_ / 16, 256, 0, stream>>>(
        embed, action, is_first, eps_prior, eps_post, idx,
        b_img_in, b_gru, ln_scale, ln_bias,
        b_img_out, b_img_dist, b_obs_out, b_obs_dist,
        gWii, gWgru, gWio, gWid, gWoo, gWod, out);
}
